// MoELayer_19258633356118
// MI455X (gfx1250) — matrix operand order repaired and checked by reading, not compiled, not run on hardware
//
#include <hip/hip_runtime.h>

// ---------------------------------------------------------------------------
// MoE top-1 layer for MI455X (gfx1250, wave32, WMMA).
// FFN matmuls run as bf16 WMMA (v_wmma_f32_16x16x32_bf16) with f32 accumulate.
// ---------------------------------------------------------------------------

typedef __attribute__((ext_vector_type(4)))  __bf16 bf16x4;
typedef __attribute__((ext_vector_type(8)))  __bf16 bf16x8;
typedef __attribute__((ext_vector_type(16))) __bf16 bf16x16;
typedef __attribute__((ext_vector_type(8)))  float  f32x8;

#define T_TOK 8192
#define D_DIM 1024
#define F_DIM 4096
#define E_NUM 8
#define LOSS_SCALE 3e-06f

// ---- workspace layout (bytes) ---------------------------------------------
#define WS_COUNTS   0                       // int[8]
#define WS_POFF     64                      // int[9]   padded offsets
#define WS_CURSOR   128                     // int[8]
#define WS_PROBSUM  192                     // float[8]
#define WS_GATE     256                     // float[T]
#define WS_EXPERT   (256 + 4 * T_TOK)       // int[T]
#define WS_TOKLIST  (256 + 8 * T_TOK)       // int[T + 16*E]
// total < 100 KB

// ---- ffn LDS layout (dynamic shared, ~173 KB of the 320 KB WGP LDS) -------
#define A_STRIDE 1032                       // 1024 + 8 pad (16B-aligned rows)
#define H_STRIDE 4104                       // 4096 + 8 pad
#define ABUF_BYTES (16 * A_STRIDE * 2)      // 33024
#define HBUF_OFF   ABUF_BYTES
#define HBUF_BYTES (16 * H_STRIDE * 2)      // 131328
#define WT_OFF     (HBUF_OFF + HBUF_BYTES)  // 164352
#define WT_BYTES   (8 * 16 * 32 * 2)        // 8192 (per-wave 16x32 bf16 tiles)
#define GBUF_OFF   (WT_OFF + WT_BYTES)      // 172544
#define TLBUF_OFF  (GBUF_OFF + 64)          // 172608
#define SMEM_BYTES (TLBUF_OFF + 64)         // 172672

// ===========================================================================
// Kernel 0: zero per-expert accumulators
// ===========================================================================
__global__ void moe_init_kernel(int* counts, float* prob_sum) {
    int i = threadIdx.x;
    if (i < E_NUM) { counts[i] = 0; prob_sum[i] = 0.0f; }
}

// ===========================================================================
// Kernel 1: router — one wave per token (wave32)
// ===========================================================================
__global__ __launch_bounds__(256)
void moe_router_kernel(const float* __restrict__ x,
                       const float* __restrict__ router_w,
                       const float* __restrict__ router_b,
                       float* __restrict__ gate,
                       int*   __restrict__ expert_id,
                       int*   __restrict__ counts,
                       float* __restrict__ prob_sum) {
    __shared__ float wsm[E_NUM * D_DIM];    // 32 KB router weights
    int tid = threadIdx.x;

    const float4* rw4 = (const float4*)router_w;
    float4* w4 = (float4*)wsm;
    for (int i = tid; i < (E_NUM * D_DIM) / 4; i += 256) w4[i] = rw4[i];
    __syncthreads();

    int wave = tid >> 5;
    int lane = tid & 31;
    int t = blockIdx.x * 8 + wave;

    const float4* xt = (const float4*)(x + (size_t)t * D_DIM);
    float4 xv[8];
#pragma unroll
    for (int j = 0; j < 8; ++j) xv[j] = xt[lane + j * 32];

    float logits[E_NUM];
#pragma unroll
    for (int e = 0; e < E_NUM; ++e) {
        float s = 0.0f;
        const float4* we = (const float4*)(wsm + e * D_DIM);
#pragma unroll
        for (int j = 0; j < 8; ++j) {
            float4 wv = we[lane + j * 32];
            s += xv[j].x * wv.x + xv[j].y * wv.y + xv[j].z * wv.z + xv[j].w * wv.w;
        }
#pragma unroll
        for (int m = 16; m >= 1; m >>= 1) s += __shfl_xor(s, m, 32);
        logits[e] = s + router_b[e];
    }

    if (lane == 0) {
        int best = 0; float bl = logits[0];
#pragma unroll
        for (int e = 1; e < E_NUM; ++e)
            if (logits[e] > bl) { bl = logits[e]; best = e; }   // first-max tie break
        float den = 0.0f;
#pragma unroll
        for (int e = 0; e < E_NUM; ++e) den += __expf(logits[e] - bl);
        float g = 1.0f / den;                                    // softmax max prob
        gate[t] = g;
        expert_id[t] = best;
        atomicAdd(&counts[best], 1);
        atomicAdd(&prob_sum[best], g);
    }
}

// ===========================================================================
// Kernel 2: padded offsets, cursors, token_list init, aux loss
// ===========================================================================
__global__ void moe_setup_kernel(const int* __restrict__ counts,
                                 const float* __restrict__ prob_sum,
                                 int* __restrict__ padded_off,
                                 int* __restrict__ cursor,
                                 int* __restrict__ token_list,
                                 float* __restrict__ loss_out) {
    __shared__ int poff[E_NUM + 1];
    if (threadIdx.x == 0) {
        int acc = 0;
        poff[0] = 0; padded_off[0] = 0;
        float loss = 0.0f;
        for (int e = 0; e < E_NUM; ++e) {
            float frac = (float)counts[e] / (float)T_TOK;
            float prob = prob_sum[e] / ((float)T_TOK * (float)T_TOK);
            loss += frac * prob;
            acc += (counts[e] + 15) & ~15;
            poff[e + 1] = acc; padded_off[e + 1] = acc;
        }
        loss_out[0] = loss * LOSS_SCALE * (float)E_NUM;
    }
    __syncthreads();
    if (threadIdx.x < E_NUM) cursor[threadIdx.x] = poff[threadIdx.x];
    for (int i = threadIdx.x; i < T_TOK + 16 * E_NUM; i += blockDim.x)
        token_list[i] = -1;
}

// ===========================================================================
// Kernel 3: scatter tokens into padded per-expert lists
// ===========================================================================
__global__ void moe_scatter_kernel(const int* __restrict__ expert_id,
                                   int* __restrict__ cursor,
                                   int* __restrict__ token_list) {
    int t = blockIdx.x * 256 + threadIdx.x;
    int e = expert_id[t];
    int pos = atomicAdd(&cursor[e], 1);
    token_list[pos] = t;
}

// ===========================================================================
// FFN fragment helpers (wave32 WMMA layouts from CDNA5 ISA §7.12.2)
// ===========================================================================
__device__ __forceinline__ bf16x16 combine16(bf16x8 lo, bf16x8 hi) {
    bf16x16 r;
#pragma unroll
    for (int i = 0; i < 8; ++i) { r[i] = lo[i]; r[i + 8] = hi[i]; }
    return r;
}

// A (16xK row-major in LDS): lane row = lane%16; v0-3 hold K=8h..8h+7, v4-7 K=+16
__device__ __forceinline__ bf16x16 load_a_frag(const __bf16* base, int stride,
                                               int k0, int lane) {
    const __bf16* p = base + (size_t)(lane & 15) * stride + k0 + 8 * (lane >> 4);
    bf16x8 lo = *(const bf16x8*)(p);
    bf16x8 hi = *(const bf16x8*)(p + 16);
    return combine16(lo, hi);
}

// B (32x16), tile staged [n][k] (k-stride 32): lane col = lane%16; v0-3 hold K=8h..8h+7, v4-7 K=+16
__device__ __forceinline__ bf16x16 load_b_frag(const __bf16* wt, int lane) {
    const __bf16* q = wt + (lane & 15) * 32 + 8 * (lane >> 4);
    bf16x8 lo = *(const bf16x8*)(q);
    bf16x8 hi = *(const bf16x8*)(q + 16);
    return combine16(lo, hi);
}

// ---- f32 -> bf16 transpose staging of one 32(K)x16(N) weight tile --------
// Lane owns a 4K x 4N micro-tile: k-consecutive values stay in-lane, so
// conversion is true v_cvt_pk pairs and stores are 4x ds_store_b64.
struct WFrag { float4 r0, r1, r2, r3; };

__device__ __forceinline__ WFrag load_w(const float* __restrict__ w,
                                        int ldw, int lane) {
    // w points at element (k0, n0); ldw = K-row stride in elements
    const float* p = w + (size_t)((lane & 7) * 4) * ldw + ((lane >> 3) * 4);
    WFrag f;
    f.r0 = *(const float4*)(p);
    f.r1 = *(const float4*)(p + ldw);
    f.r2 = *(const float4*)(p + 2 * ldw);
    f.r3 = *(const float4*)(p + 3 * ldw);
    return f;
}

__device__ __forceinline__ void store_w(const WFrag& f, __bf16* wt, int lane) {
    const int kq = (lane & 7) * 4;
    const int nq = (lane >> 3) * 4;
    bf16x4 c0 = { (__bf16)f.r0.x, (__bf16)f.r1.x, (__bf16)f.r2.x, (__bf16)f.r3.x };
    bf16x4 c1 = { (__bf16)f.r0.y, (__bf16)f.r1.y, (__bf16)f.r2.y, (__bf16)f.r3.y };
    bf16x4 c2 = { (__bf16)f.r0.z, (__bf16)f.r1.z, (__bf16)f.r2.z, (__bf16)f.r3.z };
    bf16x4 c3 = { (__bf16)f.r0.w, (__bf16)f.r1.w, (__bf16)f.r2.w, (__bf16)f.r3.w };
    __bf16* dst = wt + nq * 32 + kq;        // [n][k], 8B-aligned b64 stores
    *(bf16x4*)(dst)          = c0;
    *(bf16x4*)(dst + 32)     = c1;
    *(bf16x4*)(dst + 64)     = c2;
    *(bf16x4*)(dst + 96)     = c3;
}

// ===========================================================================
// Kernel 4: fused expert FFN. Block = 16 tokens of one expert, 8 waves.
//   phase 1: H[16,4096] = relu(bf16(X) @ bf16(W1[e]))  -> bf16 in LDS
//   phase 2: out[16,1024] = (H @ bf16(W2[e])) * gate    -> f32 global
// ===========================================================================
__global__ __launch_bounds__(256)
void moe_ffn_kernel(const float* __restrict__ x,
                    const float* __restrict__ w1,
                    const float* __restrict__ w2,
                    const int*   __restrict__ padded_off,
                    const int*   __restrict__ token_list,
                    const float* __restrict__ gate,
                    float* __restrict__ out) {
    extern __shared__ char smem[];
    __bf16* Abuf  = (__bf16*)(smem);
    __bf16* Hbuf  = (__bf16*)(smem + HBUF_OFF);
    __bf16* Wt    = (__bf16*)(smem + WT_OFF);
    float*  gbuf  = (float*)(smem + GBUF_OFF);
    int*    tlbuf = (int*)(smem + TLBUF_OFF);

    const int tid  = threadIdx.x;
    const int wave = tid >> 5;
    const int lane = tid & 31;

    const int ntiles = padded_off[E_NUM] >> 4;
    const int tile = blockIdx.x;
    if (tile >= ntiles) return;

    // which expert owns this padded tile?
    int e = 0;
    while (e < E_NUM - 1 && tile * 16 >= padded_off[e + 1]) ++e;

    if (tid < 16) {
        int t = token_list[tile * 16 + tid];
        tlbuf[tid] = t;
        gbuf[tid] = (t >= 0) ? gate[t] : 0.0f;
    }
    __syncthreads();

    // ---- stage A: 16 token rows f32 -> bf16 LDS (b64 packed stores) ----
    {
        int r = tid >> 4;                 // token row 0..15
        int cb = tid & 15;                // float4 lane within row
        int t = tlbuf[r];
        const float4* src = (const float4*)(x + (size_t)(t < 0 ? 0 : t) * D_DIM);
        float4 z4 = make_float4(0.f, 0.f, 0.f, 0.f);
#pragma unroll
        for (int j = 0; j < 16; ++j) {
            int c4 = cb + j * 16;         // 0..255
            float4 v = (t >= 0) ? src[c4] : z4;
            bf16x4 b = { (__bf16)v.x, (__bf16)v.y, (__bf16)v.z, (__bf16)v.w };
            *(bf16x4*)(Abuf + (size_t)r * A_STRIDE + c4 * 4) = b;
        }
    }
    __syncthreads();

    __bf16* wtile = Wt + wave * (16 * 32);

    // ---- phase 1: H = relu(X @ W1[e]); wave owns 512 cols of F ---------
    // Software-pipelined K loop: prefetch next weight fragment into regs
    // while the current WMMA executes.
    {
        const float* w1e = w1 + (size_t)e * D_DIM * F_DIM;
#pragma unroll 1
        for (int nc = 0; nc < 32; ++nc) {
            const int n0 = wave * 512 + nc * 16;
            const float* wp = w1e + n0;
            f32x8 acc = {0.f, 0.f, 0.f, 0.f, 0.f, 0.f, 0.f, 0.f};
            WFrag wf = load_w(wp, F_DIM, lane);                  // k0 = 0
#pragma unroll 1
            for (int k0 = 0; k0 < D_DIM; k0 += 32) {
                store_w(wf, wtile, lane);
                // prefetch next K-step (clamped pointer on final iter)
                const int kn = (k0 + 32 < D_DIM) ? (k0 + 32) : k0;
                wf = load_w(wp + (size_t)kn * F_DIM, F_DIM, lane);
                bf16x16 af = load_a_frag(Abuf, A_STRIDE, k0, lane);
                bf16x16 bf = load_b_frag(wtile, lane);
                acc = __builtin_amdgcn_wmma_f32_16x16x32_bf16(
                    false, af, false, bf, (short)0, acc, false, false);
            }
            const int hlf = lane >> 4, col = lane & 15;
#pragma unroll
            for (int r = 0; r < 8; ++r) {
                float v = acc[r];
                v = v > 0.0f ? v : 0.0f;  // relu
                Hbuf[(size_t)(r + 8 * hlf) * H_STRIDE + n0 + col] = (__bf16)v;
            }
        }
    }
    __syncthreads();

    // ---- phase 2: out = (H @ W2[e]) * gate; wave owns 128 cols of D ----
    {
        const float* w2e = w2 + (size_t)e * F_DIM * D_DIM;
        f32x8 acc2[8];
        const f32x8 z8 = {0.f, 0.f, 0.f, 0.f, 0.f, 0.f, 0.f, 0.f};
#pragma unroll
        for (int c = 0; c < 8; ++c) acc2[c] = z8;

#pragma unroll 1
        for (int k0 = 0; k0 < F_DIM; k0 += 32) {
            bf16x16 af = load_a_frag(Hbuf, H_STRIDE, k0, lane);  // reused 8x
            const float* wk = w2e + (size_t)k0 * D_DIM + wave * 128;
#pragma unroll
            for (int c = 0; c < 8; ++c) {
                store_w(load_w(wk + c * 16, D_DIM, lane), wtile, lane);
                bf16x16 bf = load_b_frag(wtile, lane);
                acc2[c] = __builtin_amdgcn_wmma_f32_16x16x32_bf16(
                    false, af, false, bf, (short)0, acc2[c], false, false);
            }
        }

        const int hlf = lane >> 4, col = lane & 15;
#pragma unroll
        for (int r = 0; r < 8; ++r) {
            const int row = r + 8 * hlf;
            const int t = tlbuf[row];
            if (t >= 0) {
                const float g = gbuf[row];
                float* orow = out + (size_t)t * D_DIM + wave * 128 + col;
#pragma unroll
                for (int c = 0; c < 8; ++c) orow[c * 16] = acc2[c][r] * g;
            }
        }
    }
}

// ===========================================================================
// host launch
// ===========================================================================
extern "C" void kernel_launch(void* const* d_in, const int* in_sizes, int n_in,
                              void* d_out, int out_size, void* d_ws, size_t ws_size,
                              hipStream_t stream) {
    (void)in_sizes; (void)n_in; (void)out_size; (void)ws_size;

    const float* x        = (const float*)d_in[0];  // [4,2048,1024]
    const float* w1       = (const float*)d_in[1];  // [8,1024,4096]
    const float* w2       = (const float*)d_in[2];  // [8,4096,1024]
    const float* router_w = (const float*)d_in[3];  // [8,1024]
    const float* router_b = (const float*)d_in[4];  // [8]
    float* out = (float*)d_out;                     // [T*D] ++ [loss]

    char* ws = (char*)d_ws;
    int*   counts     = (int*)(ws + WS_COUNTS);
    int*   padded_off = (int*)(ws + WS_POFF);
    int*   cursor     = (int*)(ws + WS_CURSOR);
    float* prob_sum   = (float*)(ws + WS_PROBSUM);
    float* gate       = (float*)(ws + WS_GATE);
    int*   expert_id  = (int*)(ws + WS_EXPERT);
    int*   token_list = (int*)(ws + WS_TOKLIST);

    moe_init_kernel<<<1, 32, 0, stream>>>(counts, prob_sum);
    moe_router_kernel<<<T_TOK / 8, 256, 0, stream>>>(
        x, router_w, router_b, gate, expert_id, counts, prob_sum);
    moe_setup_kernel<<<1, 256, 0, stream>>>(
        counts, prob_sum, padded_off, cursor, token_list,
        out + (size_t)T_TOK * D_DIM);
    moe_scatter_kernel<<<T_TOK / 256, 256, 0, stream>>>(
        expert_id, cursor, token_list);
    moe_ffn_kernel<<<T_TOK / 16 + E_NUM, 256, SMEM_BYTES, stream>>>(
        x, w1, w2, padded_off, token_list, gate, out);
}
